// MultiQueryAttention_21414706937900
// MI455X (gfx1250) — matrix operand order repaired and checked by reading, not compiled, not run on hardware
//
#include <hip/hip_runtime.h>
#include <hip/hip_bf16.h>
#include <cstdint>

// ---------------------------------------------------------------------------
// MQA forward for MI455X (gfx1250), wave32, f16 WMMA (f32 accumulate).
//   1) convert x / Wk / Wv / Wq / Wp to f16 in workspace
//   2) qh = x·Wq^T, kh = x·Wk^T, vh = x·Wv^T  (double-buffered NT WMMA GEMM)
//   3) flash-attention per (b,h,64-row q tile): S^T = K·Q^T so the softmax
//      reduces in-register (1 shuffle) and P^T lands directly in the
//      A-fragment layout for P·V; K frags + V tile software-pipelined.
//   4) out = y·Wp^T + bp  (f32 output)
// ---------------------------------------------------------------------------

#define DEV_ static __device__ __forceinline__

typedef __attribute__((ext_vector_type(4)))  _Float16 v4h;
typedef __attribute__((ext_vector_type(8)))  _Float16 v8h;
typedef __attribute__((ext_vector_type(16))) _Float16 v16h;
typedef __attribute__((ext_vector_type(8)))  float    v8f;

union V16U { v16h v; v8h h[2]; };

// A-matrix fragment (16x32 f16): lane (m=idx, half) holds K in
// {half*8 .. half*8+7} and {16+half*8 .. 16+half*8+7}  (ISA 7.12.2 table).
DEV_ v16h load_afrag(const _Float16* row, int half) {
  V16U u;
  u.h[0] = *(const v8h*)(row + half * 8);
  u.h[1] = *(const v8h*)(row + 16 + half * 8);
  return u.v;
}
// B-matrix fragment (32x16 f16): lane (n=idx, half) holds K in
// {half*8 .. half*8+7} and {16+half*8 .. 16+half*8+7} (same order as A).
DEV_ v16h load_bfrag(const _Float16* col, int half) {
  V16U u;
  u.h[0] = *(const v8h*)(col + half * 8);
  u.h[1] = *(const v8h*)(col + 16 + half * 8);
  return u.v;
}
DEV_ v8f wmma16(v16h a, v16h b, v8f c) {
  return __builtin_amdgcn_wmma_f32_16x16x32_f16(
      /*neg_a=*/false, a, /*neg_b=*/false, b,
      /*c_mod=*/(short)0, c, /*reuse_a=*/false, /*reuse_b=*/false);
}

// ---------------------------------------------------------------------------
// fp32 -> f16 conversion (vectorized x4)
// ---------------------------------------------------------------------------
__global__ __launch_bounds__(256) void cvt_f32_f16_kernel(
    const float* __restrict__ s, _Float16* __restrict__ d, int n4) {
  int i = blockIdx.x * blockDim.x + threadIdx.x;
  if (i < n4) {
    float4 f = ((const float4*)s)[i];
    v4h h;
    h[0] = (_Float16)f.x; h[1] = (_Float16)f.y;
    h[2] = (_Float16)f.z; h[3] = (_Float16)f.w;
    ((v4h*)d)[i] = h;
  }
}

// ---------------------------------------------------------------------------
// NT GEMM:  C[M,N] = A[M,K](f16) * B[N,K](f16)^T  (+bias), f32 accumulate.
// 256 threads = 8 waves; block tile 128x128x32; wave tile 64x32 (4x2 frags).
// Double-buffered LDS: global->reg prefetch overlaps the WMMA burst; one
// barrier per K-step.  K must be a multiple of 32 (true for all call sites).
// ---------------------------------------------------------------------------
template <bool BIAS, bool OUTF16>
__global__ __launch_bounds__(256) void gemm_nt_kernel(
    const _Float16* __restrict__ A, const _Float16* __restrict__ Bm,
    const float* __restrict__ bias, void* __restrict__ Cout,
    int M, int N, int K) {
  constexpr int BM = 128, BN = 128, BK = 32, LDT = BK + 8;  // +8 halves pad
  __shared__ __align__(32) _Float16 As[2][BM * LDT];
  __shared__ __align__(32) _Float16 Bs[2][BN * LDT];

  const int tid  = threadIdx.x;
  const int wid  = tid >> 5;
  const int lane = tid & 31;
  const int idx  = lane & 15;
  const int half = lane >> 4;
  const int m0 = blockIdx.y * BM;
  const int n0 = blockIdx.x * BN;
  const int wm = (wid >> 2) * 64;   // 0,64
  const int wn = (wid & 3) * 32;    // 0,32,64,96

  // two 8-half staging slots per thread per tile (512 slots of 8 halves)
  const int r0 = tid >> 2, c0 = (tid & 3) * 8, r1 = r0 + 64;
  const int ga0 = m0 + r0, ga1 = m0 + r1;
  const int gb0 = n0 + r0, gb1 = n0 + r1;

  v8f acc[4][2] = {};
  const int ntile = K / BK;

  // prologue: stage tile 0 into buffer 0
  {
    v8h t = {};
    if (ga0 < M) t = *(const v8h*)(A + (size_t)ga0 * K + c0);
    *(v8h*)(&As[0][r0 * LDT + c0]) = t;
    t = (v8h){};
    if (ga1 < M) t = *(const v8h*)(A + (size_t)ga1 * K + c0);
    *(v8h*)(&As[0][r1 * LDT + c0]) = t;
    t = (v8h){};
    if (gb0 < N) t = *(const v8h*)(Bm + (size_t)gb0 * K + c0);
    *(v8h*)(&Bs[0][r0 * LDT + c0]) = t;
    t = (v8h){};
    if (gb1 < N) t = *(const v8h*)(Bm + (size_t)gb1 * K + c0);
    *(v8h*)(&Bs[0][r1 * LDT + c0]) = t;
  }
  __syncthreads();

  int cur = 0;
  for (int kt = 0; kt < ntile; ++kt) {
    const bool hn = (kt + 1 < ntile);
    v8h a0v = {}, a1v = {}, b0v = {}, b1v = {};
    if (hn) {
      const int k0n = (kt + 1) * BK;
      if (ga0 < M) a0v = *(const v8h*)(A + (size_t)ga0 * K + k0n + c0);
      if (ga1 < M) a1v = *(const v8h*)(A + (size_t)ga1 * K + k0n + c0);
      if (gb0 < N) b0v = *(const v8h*)(Bm + (size_t)gb0 * K + k0n + c0);
      if (gb1 < N) b1v = *(const v8h*)(Bm + (size_t)gb1 * K + k0n + c0);
      if (kt + 2 < ntile) {  // L2 prefetch two tiles ahead (global_prefetch_b8)
        const int k0p = (kt + 2) * BK;
        if (ga0 < M) __builtin_prefetch(A + (size_t)ga0 * K + k0p + c0, 0, 1);
        if (gb0 < N) __builtin_prefetch(Bm + (size_t)gb0 * K + k0p + c0, 0, 1);
      }
    }

    v16h af[4], bf[2];
    #pragma unroll
    for (int i = 0; i < 4; ++i)
      af[i] = load_afrag(&As[cur][(wm + i * 16 + idx) * LDT], half);
    #pragma unroll
    for (int j = 0; j < 2; ++j)
      bf[j] = load_bfrag(&Bs[cur][(wn + j * 16 + idx) * LDT], half);
    #pragma unroll
    for (int i = 0; i < 4; ++i)
      #pragma unroll
      for (int j = 0; j < 2; ++j)
        acc[i][j] = wmma16(af[i], bf[j], acc[i][j]);

    if (hn) {
      const int nxt = cur ^ 1;
      *(v8h*)(&As[nxt][r0 * LDT + c0]) = a0v;
      *(v8h*)(&As[nxt][r1 * LDT + c0]) = a1v;
      *(v8h*)(&Bs[nxt][r0 * LDT + c0]) = b0v;
      *(v8h*)(&Bs[nxt][r1 * LDT + c0]) = b1v;
      __syncthreads();
      cur = nxt;
    }
  }

  // C/D layout: lane (idx, half), vgpr e -> (row = half*8+e, col = idx)
  #pragma unroll
  for (int i = 0; i < 4; ++i) {
    #pragma unroll
    for (int j = 0; j < 2; ++j) {
      int col = n0 + wn + j * 16 + idx;
      #pragma unroll
      for (int e = 0; e < 8; ++e) {
        int row = m0 + wm + i * 16 + half * 8 + e;
        if (row < M && col < N) {
          float v = acc[i][j][e];
          if (BIAS) v += bias[col];
          if (OUTF16) ((_Float16*)Cout)[(size_t)row * N + col] = (_Float16)v;
          else        ((float*)Cout)[(size_t)row * N + col] = v;
        }
      }
    }
  }
}

// ---------------------------------------------------------------------------
// Flash attention (causal, multi-query: 1 shared K/V head).
// Block = 128 threads = 4 waves; one block per (b, h, 64-row q tile).
// Wave w owns 16 q rows; 32 keys per iteration.
//
// S^T = K·Q^T: each lane gets 16 of the 32 key scores of ONE q row
// (col = idx) -> row stats need one shfl_xor(16); per-lane register order
// (keys half*8+e | 16+half*8+e) is exactly the A-fragment of P[16x32].
// K fragments and the V^T LDS tile are software-pipelined (prefetch next
// tile during the current tile's softmax); 1/sqrt(hs) folded into Q (exact,
// power of two).  Masking only on diagonal-crossing tiles (wave-uniform).
// ---------------------------------------------------------------------------
__global__ __launch_bounds__(128) void mqa_attn_kernel(
    const _Float16* __restrict__ qh, const _Float16* __restrict__ kh,
    const _Float16* __restrict__ vh, _Float16* __restrict__ yh) {
  constexpr int T = 2048, C = 1024, HS = 64;
  constexpr int VS = 32 + 8;   // V^T LDS row stride (halves)
  __shared__ __align__(32) _Float16 vt[2][HS * VS];  // V^T [64 hs][32 keys] x2

  const int tid  = threadIdx.x;
  const int w    = tid >> 5;
  const int lane = tid & 31;
  const int idx  = lane & 15;
  const int half = lane >> 4;
  const int qt = blockIdx.x, h = blockIdx.y, b = blockIdx.z;
  const int qbase = qt * 64;
  const int qoff  = qbase + w * 16;

  // torch raw reshape: Q(b,h) is contiguous [T,64] at b*T*C + h*T*HS
  const _Float16* Q  = qh + (size_t)b * T * C + (size_t)h * T * HS;
  const _Float16* Kb = kh + (size_t)b * T * HS;
  const _Float16* Vb = vh + (size_t)b * T * HS;

  // Q as B-fragments of S^T = K·Q^T; fold in hs^-0.5 = 0.125 (exact in f16)
  const _Float16* qrow = Q + (size_t)(qoff + idx) * HS;
  v16h bq0 = load_bfrag(qrow, half);        // hs 0..31
  v16h bq1 = load_bfrag(qrow + 32, half);   // hs 32..63
  {
    const _Float16 qs = (_Float16)0.125f;
    #pragma unroll
    for (int e2 = 0; e2 < 16; ++e2) { bq0[e2] *= qs; bq1[e2] *= qs; }
  }

  v8f o[4] = {};                   // O rows = q (half*8+e), cols = hs (idx)
  float mi = -INFINITY, li = 0.f;  // stats for q row (qoff + idx)

  const int nkb = 2 * (qt + 1);    // keys [0, qbase+64), uniform per block
  const int qr  = qoff + idx;

  // V staging slots: 4 x v4h per thread per tile
  const int vkey = tid >> 4;        // 0..7 (+8*j)
  const int vcg  = (tid & 15) * 4;  // hs group of 4

  // K A-fragment loader (4 frags: 2 key subtiles x 2 hs slabs)
  auto loadK = [&](int kbase, v16h* ak) {
    const _Float16* kr0 = Kb + (size_t)(kbase + idx) * HS;
    const _Float16* kr1 = Kb + (size_t)(kbase + 16 + idx) * HS;
    ak[0] = load_afrag(kr0, half);
    ak[1] = load_afrag(kr0 + 32, half);
    ak[2] = load_afrag(kr1, half);
    ak[3] = load_afrag(kr1 + 32, half);
  };
  auto loadV = [&](int kbase, v4h* vr) {
    #pragma unroll
    for (int j = 0; j < 4; ++j)
      vr[j] = *(const v4h*)(Vb + (size_t)(kbase + vkey + 8 * j) * HS + vcg);
  };
  auto storeV = [&](int buf, const v4h* vr) {
    #pragma unroll
    for (int j = 0; j < 4; ++j)
      #pragma unroll
      for (int jj = 0; jj < 4; ++jj)
        vt[buf][(vcg + jj) * VS + vkey + 8 * j] = vr[j][jj];
  };

  // prologue: V tile 0 -> vt[0]; K frags for tile 0 in flight
  v16h ak[4];
  loadK(0, ak);
  {
    v4h vr[4];
    loadV(0, vr);
    storeV(0, vr);
  }
  __syncthreads();

  int cur = 0;
  for (int kb = 0; kb < nkb; ++kb) {
    const int kbase = kb * 32;
    const bool hn = (kb + 1 < nkb);

    // prefetch next tile (K frags to regs, V to regs) during this tile's math
    v16h akn[4];
    v4h  vrn[4];
    if (hn) {
      loadK(kbase + 32, akn);
      loadV(kbase + 32, vrn);
    }

    // S^T = K·Q^T
    v8f z = {};
    v8f st0 = wmma16(ak[0], bq0, z);
    st0 = wmma16(ak[1], bq1, st0);    // key rows kbase..kbase+15
    v8f st1 = wmma16(ak[2], bq0, z);
    st1 = wmma16(ak[3], bq1, st1);    // key rows kbase+16..kbase+31

    float x0[8], x1[8];
    #pragma unroll
    for (int e = 0; e < 8; ++e) { x0[e] = st0[e]; x1[e] = st1[e]; }
    if (kbase + 31 > qoff) {  // wave-uniform: tile crosses this wave's diagonal
      #pragma unroll
      for (int e = 0; e < 8; ++e) {
        const int kr = kbase + half * 8 + e;
        if (kr      > qr) x0[e] = -INFINITY;
        if (kr + 16 > qr) x1[e] = -INFINITY;
      }
    }

    // row max: in-register over 16 values + 1 cross-half shuffle
    float mx = x0[0];
    #pragma unroll
    for (int e = 1; e < 8; ++e) mx = fmaxf(mx, x0[e]);
    #pragma unroll
    for (int e = 0; e < 8; ++e) mx = fmaxf(mx, x1[e]);
    mx = fmaxf(mx, __shfl_xor(mx, 16, 32));
    const float mnew = fmaxf(mi, mx);
    const float a = __expf(mi - mnew);
    mi = mnew;

    // P^T values: exp(x - mnew); pack straight into the A-fragment of P
    V16U pu;
    float rs = 0.f;
    #pragma unroll
    for (int e = 0; e < 8; ++e) {
      float p0 = __expf(x0[e] - mnew);
      float p1 = __expf(x1[e] - mnew);
      rs += p0 + p1;
      pu.h[0][e] = (_Float16)p0;   // keys half*8+e
      pu.h[1][e] = (_Float16)p1;   // keys 16+half*8+e
    }
    rs += __shfl_xor(rs, 16, 32);
    li = li * a + rs;

    // redistribute alpha to O's row layout (row half*8+e lives in that lane)
    #pragma unroll
    for (int e = 0; e < 8; ++e) {
      const float ar = __shfl(a, half * 8 + e, 32);
      #pragma unroll
      for (int nt = 0; nt < 4; ++nt) o[nt][e] *= ar;
    }

    // O += P · V   (A = pu.v is P[16q x 32k]; B = V^T columns from LDS)
    #pragma unroll
    for (int nt = 0; nt < 4; ++nt) {
      v16h bv = load_bfrag(&vt[cur][(nt * 16 + idx) * VS], half);
      o[nt] = wmma16(pu.v, bv, o[nt]);
    }

    if (hn) {
      storeV(cur ^ 1, vrn);
      __syncthreads();
      cur ^= 1;
      #pragma unroll
      for (int q4 = 0; q4 < 4; ++q4) ak[q4] = akn[q4];
    }
  }

  // normalize (1/l for row half*8+e lives in lane half*8+e) and store y
  const float linv = 1.0f / li;
  float inv[8];
  #pragma unroll
  for (int e = 0; e < 8; ++e) inv[e] = __shfl(linv, half * 8 + e, 32);

  const size_t outb = (size_t)b * T * C;
  #pragma unroll
  for (int nt = 0; nt < 4; ++nt) {
    const int col = h * HS + nt * 16 + idx;
    #pragma unroll
    for (int e = 0; e < 8; ++e) {
      const int t2 = qoff + half * 8 + e;
      yh[outb + (size_t)t2 * C + col] = (_Float16)(o[nt][e] * inv[e]);
    }
  }
}

// ---------------------------------------------------------------------------
extern "C" void kernel_launch(void* const* d_in, const int* in_sizes, int n_in,
                              void* d_out, int out_size, void* d_ws, size_t ws_size,
                              hipStream_t stream) {
  (void)in_sizes; (void)n_in; (void)out_size; (void)ws_size;
  constexpr int B = 2, T = 2048, C = 1024, HS = 64, H = 16;
  constexpr int BT = B * T;  // 4096

  const float* x  = (const float*)d_in[0];
  const float* Wk = (const float*)d_in[1];
  const float* Wv = (const float*)d_in[2];
  const float* Wq = (const float*)d_in[3];
  const float* Wp = (const float*)d_in[4];
  const float* bp = (const float*)d_in[5];

  // workspace carve (f16 buffers), 256B aligned; total ~30 MB
  size_t off = 0;
  auto carve = [&](size_t halves) {
    void* p = (char*)d_ws + off;
    off += (halves * sizeof(_Float16) + 255) & ~(size_t)255;
    return (_Float16*)p;
  };
  _Float16* xh  = carve((size_t)BT * C);
  _Float16* qh  = carve((size_t)BT * C);
  _Float16* yh  = carve((size_t)BT * C);
  _Float16* kh  = carve((size_t)BT * HS);
  _Float16* vh  = carve((size_t)BT * HS);
  _Float16* wqh = carve((size_t)C * C);
  _Float16* wph = carve((size_t)C * C);
  _Float16* wkh = carve((size_t)HS * C);
  _Float16* wvh = carve((size_t)HS * C);

  auto cvt = [&](const float* s, _Float16* d, size_t n) {
    int n4 = (int)(n / 4);
    cvt_f32_f16_kernel<<<(n4 + 255) / 256, 256, 0, stream>>>(s, d, n4);
  };
  cvt(x,  xh,  (size_t)BT * C);
  cvt(Wq, wqh, (size_t)C * C);
  cvt(Wp, wph, (size_t)C * C);
  cvt(Wk, wkh, (size_t)HS * C);
  cvt(Wv, wvh, (size_t)HS * C);

  const dim3 blk(256);
  dim3 gq((C + 127) / 128, (BT + 127) / 128);    // (8, 32)
  dim3 gs((HS + 127) / 128, (BT + 127) / 128);   // (1, 32)
  gemm_nt_kernel<false, true><<<gq, blk, 0, stream>>>(xh, wqh, nullptr, qh, BT, C, C);
  gemm_nt_kernel<false, true><<<gs, blk, 0, stream>>>(xh, wkh, nullptr, kh, BT, HS, C);
  gemm_nt_kernel<false, true><<<gs, blk, 0, stream>>>(xh, wvh, nullptr, vh, BT, HS, C);

  mqa_attn_kernel<<<dim3(T / 64, H, B), dim3(128), 0, stream>>>(qh, kh, vh, yh);

  gemm_nt_kernel<true, false><<<gq, blk, 0, stream>>>(yh, wph, bp, d_out, BT, C, C);
}
